// SparseCrossAttnAdapter_76175539961968
// MI455X (gfx1250) — matrix operand order repaired and checked by reading, not compiled, not run on hardware
//
#include <hip/hip_runtime.h>
#include <stdint.h>

#define HIDDEN   1024
#define HEADS    16
#define HEAD_DIM 64
#define LIMG     4096
#define LCOND    2048
#define BATCH    2
#define BK       64

typedef __attribute__((ext_vector_type(16))) __bf16 v16bf;
typedef __attribute__((ext_vector_type(8)))  __bf16 v8bf;
typedef __attribute__((ext_vector_type(8)))  float  v8f;

// ---------------- helpers ----------------

__device__ __forceinline__ __bf16 f2bf(float f) {
    union { float f; uint32_t u; } a; a.f = f;
    uint32_t r = a.u + 0x7FFFu + ((a.u >> 16) & 1u);   // RNE
    union { uint16_t s; __bf16 b; } o; o.s = (uint16_t)(r >> 16);
    return o.b;
}

__device__ __forceinline__ v16bf ld_frag(const __bf16* p0, const __bf16* p1) {
    union { v16bf v; v8bf h[2]; } u;
    u.h[0] = *(const v8bf*)p0;
    u.h[1] = *(const v8bf*)p1;
    return u.v;
}

__device__ __forceinline__ v8f wmma_bf16(v16bf a, v16bf b, v8f c) {
    return __builtin_amdgcn_wmma_f32_16x16x32_bf16(false, a, false, b, (short)0, c, false, false);
}

// CDNA5 async DMA: global -> LDS, 16 bytes per lane, tracked by ASYNCcnt.
// LDS byte address = low 32 bits of the generic shared pointer (aperture rule).
__device__ __forceinline__ void async_g2l_b128(void* lds, const void* g) {
    uint32_t l32 = (uint32_t)(uintptr_t)lds;
    asm volatile("global_load_async_to_lds_b128 %0, %1, off"
                 :: "v"(l32), "v"(g) : "memory");
}
__device__ __forceinline__ void wait_async_le8() {
    asm volatile("s_wait_asynccnt 0x8" ::: "memory");
}
__device__ __forceinline__ void wait_async_0() {
    asm volatile("s_wait_asynccnt 0x0" ::: "memory");
}

// ---------------- elementwise kernels ----------------

__global__ void cast_w_kernel(const float* __restrict__ in, __bf16* __restrict__ out, int n) {
    int i0 = (blockIdx.x * 256 + threadIdx.x) * 4;
    if (i0 + 3 < n) {
        float4 v = *(const float4*)(in + i0);
        out[i0 + 0] = f2bf(v.x); out[i0 + 1] = f2bf(v.y);
        out[i0 + 2] = f2bf(v.z); out[i0 + 3] = f2bf(v.w);
    }
}

__global__ void rmsnorm_cast_kernel(const float* __restrict__ X, const float* __restrict__ w,
                                    __bf16* __restrict__ out) {
    int row = blockIdx.x;
    int t = threadIdx.x;
    const float* xr = X + (size_t)row * HIDDEN;
    float4 v = ((const float4*)xr)[t];
    float ss = v.x * v.x + v.y * v.y + v.z * v.z + v.w * v.w;
    #pragma unroll
    for (int off = 16; off > 0; off >>= 1) ss += __shfl_xor(ss, off, 32);
    __shared__ float red[8];
    if ((t & 31) == 0) red[t >> 5] = ss;
    __syncthreads();
    float tot = 0.f;
    #pragma unroll
    for (int i = 0; i < 8; ++i) tot += red[i];
    float inv = rsqrtf(tot * (1.0f / HIDDEN) + 1e-6f);
    int c = t * 4;
    float4 ww = *(const float4*)(w + c);
    __bf16* orow = out + (size_t)row * HIDDEN + c;
    orow[0] = f2bf(v.x * inv * ww.x);
    orow[1] = f2bf(v.y * inv * ww.y);
    orow[2] = f2bf(v.z * inv * ww.z);
    orow[3] = f2bf(v.w * inv * ww.w);
}

__global__ void rope_pack_kernel(const float* __restrict__ Y,
                                 const float* __restrict__ cosT, const float* __restrict__ sinT,
                                 const float* __restrict__ hw,
                                 __bf16* __restrict__ out, int L) {
    int row = blockIdx.x;            // b*L + l
    int b = row / L, l = row - b * L;
    int t = threadIdx.x;
    int h = t >> 4;
    int d0 = (t & 15) * 4;
    const float* xr = Y + (size_t)row * HIDDEN + h * HEAD_DIM + d0;
    float x0 = xr[0], x1 = xr[1], x2 = xr[2], x3 = xr[3];
    float ss = x0 * x0 + x1 * x1 + x2 * x2 + x3 * x3;
    #pragma unroll
    for (int off = 1; off < 16; off <<= 1) ss += __shfl_xor(ss, off, 16);
    float inv = rsqrtf(ss * (1.0f / HEAD_DIM) + 1e-6f);
    x0 *= inv * hw[d0 + 0]; x1 *= inv * hw[d0 + 1];
    x2 *= inv * hw[d0 + 2]; x3 *= inv * hw[d0 + 3];
    const float* cr = cosT + (size_t)l * HEAD_DIM + d0;
    const float* sr = sinT + (size_t)l * HEAD_DIM + d0;
    float y0 = x0 * cr[0] - x1 * sr[0];
    float y1 = x1 * cr[1] + x0 * sr[1];
    float y2 = x2 * cr[2] - x3 * sr[2];
    float y3 = x3 * cr[3] + x2 * sr[3];
    __bf16* orow = out + (((size_t)(b * HEADS + h) * L) + l) * HEAD_DIM + d0;
    orow[0] = f2bf(y0); orow[1] = f2bf(y1); orow[2] = f2bf(y2); orow[3] = f2bf(y3);
}

__global__ void pack_v_kernel(const float* __restrict__ Yv, __bf16* __restrict__ Vt) {
    int l = blockIdx.x * 256 + threadIdx.x;
    int rowo = blockIdx.y;                       // b*1024 + h*64 + d
    int b = rowo >> 10;
    int hd = rowo & 1023;
    int h = hd >> 6, d = hd & 63;
    float v = Yv[(((size_t)(b * LCOND + l) * HEADS) + h) * HEAD_DIM + d];
    Vt[(size_t)rowo * LCOND + l] = f2bf(v);
}

// ---------------- WMMA GEMM: Y = A[M,1024](bf16) x W[1024,1024]^T(bf16) + bias ----------------
// mode 0: out f32 = result + bias.   mode 1: out = resid + alpha * (result + bias)
// Double-buffered LDS tiles filled by async global->LDS DMA.

__global__ __launch_bounds__(256)
void gemm_bf16_kernel(const __bf16* __restrict__ A, const __bf16* __restrict__ W,
                      const float* __restrict__ bias, float* __restrict__ out,
                      const float* __restrict__ resid, const float* __restrict__ alphap,
                      int mode) {
    __shared__ __align__(16) __bf16 As[2][128 * BK];
    __shared__ __align__(16) __bf16 Bs[2][128 * BK];
    int tid = threadIdx.x;
    int wid = tid >> 5, lane = tid & 31, l = lane & 15, hi = lane >> 4;
    int wm = wid & 3, wn = wid >> 2;
    int m0 = blockIdx.y * 128, n0 = blockIdx.x * 128;

    v8f acc[2][4];
    #pragma unroll
    for (int rt = 0; rt < 2; ++rt)
        #pragma unroll
        for (int ct = 0; ct < 4; ++ct) acc[rt][ct] = (v8f)(0.f);

    // async-issue one (A,B) K-tile pair into buffer `buf`: 8 DMA instrs / wave
    auto issue = [&](int buf, int k0) {
        #pragma unroll
        for (int it = 0; it < 4; ++it) {
            int c = tid + it * 256;            // 0..1023 chunks of 16B
            int row = c >> 3, ko = (c & 7) * 8;
            async_g2l_b128(&As[buf][row * BK + ko],
                           &A[(size_t)(m0 + row) * HIDDEN + k0 + ko]);
            async_g2l_b128(&Bs[buf][row * BK + ko],
                           &W[(size_t)(n0 + row) * HIDDEN + k0 + ko]);
        }
    };

    issue(0, 0);
    const int T = HIDDEN / BK;                 // 16
    for (int t = 0; t < T; ++t) {
        int buf = t & 1;
        if (t + 1 < T) { issue(buf ^ 1, (t + 1) * BK); wait_async_le8(); }
        else           { wait_async_0(); }
        __syncthreads();                       // tile t visible to all waves

        #pragma unroll
        for (int ks = 0; ks < 2; ++ks) {
            v16bf af[2], bfm[4];
            #pragma unroll
            for (int rt = 0; rt < 2; ++rt) {
                const __bf16* base = &As[buf][(wm * 32 + rt * 16 + l) * BK + ks * 32];
                af[rt] = ld_frag(base + (hi ? 8 : 0), base + (hi ? 24 : 16));
            }
            #pragma unroll
            for (int ct = 0; ct < 4; ++ct) {
                const __bf16* base = &Bs[buf][(wn * 64 + ct * 16 + l) * BK + ks * 32];
                bfm[ct] = ld_frag(base + (hi ? 8 : 0), base + (hi ? 24 : 16));
            }
            #pragma unroll
            for (int rt = 0; rt < 2; ++rt)
                #pragma unroll
                for (int ct = 0; ct < 4; ++ct)
                    acc[rt][ct] = wmma_bf16(af[rt], bfm[ct], acc[rt][ct]);
        }
        __syncthreads();                       // readers done before buffer reuse
    }

    float alpha = (mode == 1) ? alphap[0] : 0.f;
    #pragma unroll
    for (int rt = 0; rt < 2; ++rt)
        #pragma unroll
        for (int ct = 0; ct < 4; ++ct) {
            int n = n0 + wn * 64 + ct * 16 + l;
            float bn = bias[n];
            #pragma unroll
            for (int i = 0; i < 8; ++i) {
                int m = m0 + wm * 32 + rt * 16 + i + hi * 8;
                float v = acc[rt][ct][i] + bn;
                size_t idx = (size_t)m * HIDDEN + n;
                if (mode == 1) v = resid[idx] + alpha * v;
                out[idx] = v;
            }
        }
}

// ---------------- flash cross-attention ----------------
// Q bf16 [B,H,LIMG,64], K bf16 [B,H,LCOND,64], Vt bf16 [B,H,64,LCOND]
// O bf16 [B, LIMG, H*64]

__global__ __launch_bounds__(256)
void flash_attn_kernel(const __bf16* __restrict__ Q, const __bf16* __restrict__ K,
                       const __bf16* __restrict__ Vt, __bf16* __restrict__ O) {
    __shared__ __align__(16) __bf16 Ks[2][128 * 64];
    __shared__ __align__(16) __bf16 Vs[2][64 * 128];
    __shared__ __align__(16) __bf16 Ps[8 * 16 * 128];

    int tid = threadIdx.x;
    int wid = tid >> 5, lane = tid & 31, l = lane & 15, hi = lane >> 4;
    int bh = blockIdx.y;
    int q0 = blockIdx.x * 128;

    const __bf16* Qb = Q + (size_t)bh * LIMG * HEAD_DIM;
    const __bf16* Kb = K + (size_t)bh * LCOND * HEAD_DIM;
    const __bf16* Vb = Vt + (size_t)bh * HEAD_DIM * LCOND;

    const __bf16* qrow = Qb + (size_t)(q0 + wid * 16 + l) * HEAD_DIM;
    v16bf qa[2];
    #pragma unroll
    for (int s = 0; s < 2; ++s)
        qa[s] = ld_frag(qrow + s * 32 + (hi ? 8 : 0), qrow + s * 32 + (hi ? 24 : 16));

    float m_[8], l_[8];
    v8f acc[4];
    #pragma unroll
    for (int i = 0; i < 8; ++i) { m_[i] = -1e30f; l_[i] = 0.f; }
    #pragma unroll
    for (int ct = 0; ct < 4; ++ct) acc[ct] = (v8f)(0.f);

    __bf16* Pw = &Ps[wid * 16 * 128];

    // async-issue one (K, Vt) kv-tile pair: 8 DMA instrs / wave
    auto issue = [&](int buf, int kt) {
        #pragma unroll
        for (int it = 0; it < 4; ++it) {
            int c = tid + it * 256;            // 0..1023
            async_g2l_b128(&Ks[buf][c * 8], &Kb[(size_t)kt * HEAD_DIM + c * 8]);
            int vr = c >> 4, vo = (c & 15) * 8;
            async_g2l_b128(&Vs[buf][vr * 128 + vo], &Vb[(size_t)vr * LCOND + kt + vo]);
        }
    };

    issue(0, 0);
    const int T = LCOND / 128;                 // 16
    for (int t = 0; t < T; ++t) {
        int buf = t & 1;
        if (t + 1 < T) { issue(buf ^ 1, (t + 1) * 128); wait_async_le8(); }
        else           { wait_async_0(); }
        __syncthreads();

        // S = (Q K^T) * scale  -- 16x128 strip per wave
        v8f S[8];
        #pragma unroll
        for (int n = 0; n < 8; ++n) {
            v8f s = (v8f)(0.f);
            #pragma unroll
            for (int ks = 0; ks < 2; ++ks) {
                const __bf16* kp = &Ks[buf][(n * 16 + l) * 64 + ks * 32];
                s = wmma_bf16(qa[ks], ld_frag(kp + (hi ? 8 : 0), kp + (hi ? 24 : 16)), s);
            }
            S[n] = s;
        }

        float mn[8], corr[8], rs[8];
        #pragma unroll
        for (int i = 0; i < 8; ++i) {
            float tmax = -1e30f;
            #pragma unroll
            for (int n = 0; n < 8; ++n) { S[n][i] *= 0.125f; tmax = fmaxf(tmax, S[n][i]); }
            mn[i] = tmax;
        }
        #pragma unroll
        for (int i = 0; i < 8; ++i)
            #pragma unroll
            for (int off = 1; off < 16; off <<= 1)
                mn[i] = fmaxf(mn[i], __shfl_xor(mn[i], off, 16));
        #pragma unroll
        for (int i = 0; i < 8; ++i) {
            float mnew = fmaxf(m_[i], mn[i]);
            corr[i] = __expf(m_[i] - mnew);
            m_[i] = mnew; rs[i] = 0.f;
        }
        #pragma unroll
        for (int n = 0; n < 8; ++n)
            #pragma unroll
            for (int i = 0; i < 8; ++i) {
                float p = __expf(S[n][i] - m_[i]);
                rs[i] += p;
                Pw[(i + hi * 8) * 128 + n * 16 + l] = f2bf(p);
            }
        #pragma unroll
        for (int i = 0; i < 8; ++i)
            #pragma unroll
            for (int off = 1; off < 16; off <<= 1)
                rs[i] += __shfl_xor(rs[i], off, 16);
        #pragma unroll
        for (int i = 0; i < 8; ++i) l_[i] = l_[i] * corr[i] + rs[i];
        #pragma unroll
        for (int ct = 0; ct < 4; ++ct)
            #pragma unroll
            for (int i = 0; i < 8; ++i) acc[ct][i] *= corr[i];

        // acc += P @ V
        #pragma unroll
        for (int ks = 0; ks < 4; ++ks) {
            const __bf16* pr = Pw + l * 128 + ks * 32;
            v16bf a = ld_frag(pr + (hi ? 8 : 0), pr + (hi ? 24 : 16));
            #pragma unroll
            for (int ct = 0; ct < 4; ++ct) {
                const __bf16* vp = &Vs[buf][(ct * 16 + l) * 128 + ks * 32];
                acc[ct] = wmma_bf16(a, ld_frag(vp + (hi ? 8 : 0), vp + (hi ? 24 : 16)), acc[ct]);
            }
        }
        __syncthreads();
    }

    int b = bh >> 4, h = bh & 15;
    #pragma unroll
    for (int ct = 0; ct < 4; ++ct)
        #pragma unroll
        for (int i = 0; i < 8; ++i) {
            int row = q0 + wid * 16 + i + hi * 8;
            float inv = 1.0f / l_[i];
            O[((size_t)(b * LIMG + row)) * HIDDEN + h * HEAD_DIM + ct * 16 + l] =
                f2bf(acc[ct][i] * inv);
        }
}

// ---------------- launch ----------------

extern "C" void kernel_launch(void* const* d_in, const int* in_sizes, int n_in,
                              void* d_out, int out_size, void* d_ws, size_t ws_size,
                              hipStream_t stream) {
    const float* img      = (const float*)d_in[0];
    const float* cond     = (const float*)d_in[1];
    const float* img_cos  = (const float*)d_in[2];
    const float* img_sin  = (const float*)d_in[3];
    const float* cond_cos = (const float*)d_in[4];
    const float* cond_sin = (const float*)d_in[5];
    const float* q_norm_w = (const float*)d_in[6];
    const float* kv_norm_w= (const float*)d_in[7];
    const float* q_head_w = (const float*)d_in[8];
    const float* k_head_w = (const float*)d_in[9];
    const float* qW = (const float*)d_in[10]; const float* qb = (const float*)d_in[11];
    const float* kW = (const float*)d_in[12]; const float* kb = (const float*)d_in[13];
    const float* vW = (const float*)d_in[14]; const float* vb = (const float*)d_in[15];
    const float* oW = (const float*)d_in[16]; const float* ob = (const float*)d_in[17];
    const float* alpha = (const float*)d_in[18];

    char* ws = (char*)d_ws;
    size_t off = 0;
    auto alloc = [&](size_t bytes) -> void* {
        off = (off + 255) & ~(size_t)255;
        void* p = ws + off;
        off += bytes;
        return p;
    };
    const size_t WN = (size_t)HIDDEN * HIDDEN;
    __bf16* Wq = (__bf16*)alloc(WN * 2);
    __bf16* Wk = (__bf16*)alloc(WN * 2);
    __bf16* Wv = (__bf16*)alloc(WN * 2);
    __bf16* Wo = (__bf16*)alloc(WN * 2);
    __bf16* Xq  = (__bf16*)alloc((size_t)BATCH * LIMG  * HIDDEN * 2);
    __bf16* Xkv = (__bf16*)alloc((size_t)BATCH * LCOND * HIDDEN * 2);
    float*  Yq  = (float*)alloc((size_t)BATCH * LIMG  * HIDDEN * 4);
    float*  Yk  = (float*)alloc((size_t)BATCH * LCOND * HIDDEN * 4);
    float*  Yv  = (float*)alloc((size_t)BATCH * LCOND * HIDDEN * 4);
    __bf16* Qb  = (__bf16*)alloc((size_t)BATCH * HEADS * LIMG  * HEAD_DIM * 2);
    __bf16* Kb  = (__bf16*)alloc((size_t)BATCH * HEADS * LCOND * HEAD_DIM * 2);
    __bf16* Vtb = (__bf16*)alloc((size_t)BATCH * HEADS * HEAD_DIM * LCOND * 2);
    __bf16* Ob  = (__bf16*)alloc((size_t)BATCH * LIMG * HIDDEN * 2);

    cast_w_kernel<<<1024, 256, 0, stream>>>(qW, Wq, (int)WN);
    cast_w_kernel<<<1024, 256, 0, stream>>>(kW, Wk, (int)WN);
    cast_w_kernel<<<1024, 256, 0, stream>>>(vW, Wv, (int)WN);
    cast_w_kernel<<<1024, 256, 0, stream>>>(oW, Wo, (int)WN);

    rmsnorm_cast_kernel<<<BATCH * LIMG,  256, 0, stream>>>(img,  q_norm_w,  Xq);
    rmsnorm_cast_kernel<<<BATCH * LCOND, 256, 0, stream>>>(cond, kv_norm_w, Xkv);

    gemm_bf16_kernel<<<dim3(8, (BATCH * LIMG) / 128),  256, 0, stream>>>(Xq,  Wq, qb, Yq, nullptr, nullptr, 0);
    gemm_bf16_kernel<<<dim3(8, (BATCH * LCOND) / 128), 256, 0, stream>>>(Xkv, Wk, kb, Yk, nullptr, nullptr, 0);
    gemm_bf16_kernel<<<dim3(8, (BATCH * LCOND) / 128), 256, 0, stream>>>(Xkv, Wv, vb, Yv, nullptr, nullptr, 0);

    rope_pack_kernel<<<BATCH * LIMG,  256, 0, stream>>>(Yq, img_cos,  img_sin,  q_head_w, Qb, LIMG);
    rope_pack_kernel<<<BATCH * LCOND, 256, 0, stream>>>(Yk, cond_cos, cond_sin, k_head_w, Kb, LCOND);
    pack_v_kernel<<<dim3(LCOND / 256, BATCH * HEADS * HEAD_DIM), 256, 0, stream>>>(Yv, Vtb);

    flash_attn_kernel<<<dim3(LIMG / 128, BATCH * HEADS), 256, 0, stream>>>(Qb, Kb, Vtb, Ob);

    gemm_bf16_kernel<<<dim3(8, (BATCH * LIMG) / 128), 256, 0, stream>>>(Ob, Wo, ob, (float*)d_out, img, alpha, 1);
}
